// SelfAttention_13151189860389
// MI455X (gfx1250) — matrix operand order repaired and checked by reading, not compiled, not run on hardware
//
#include <hip/hip_runtime.h>
#include <hip/hip_bf16.h>

// ---------------------------------------------------------------------------
// Gated linear attention for MI455X (gfx1250, wave32, WMMA).
//   out = (sigmoid(xWg+bg) * attn) Wo + bo
//   attn = phi(q) @ (phi(k)^T v) per head, phi = elu+1
// Matmuls via v_wmma_f32_16x16x32_f16; tiles streamed with double-buffered
// async global->LDS copies (ASYNCcnt) when the toolchain exposes them.
// ---------------------------------------------------------------------------

typedef __attribute__((ext_vector_type(16))) _Float16 v16h;
typedef __attribute__((ext_vector_type(8)))  float    v8f;
typedef int v4i_gcc __attribute__((__vector_size__(4 * sizeof(int))));

#define WMMA_F16(a, b, c) \
    __builtin_amdgcn_wmma_f32_16x16x32_f16(false, (a), false, (b), (short)0, (c), false, false)

#if defined(__AMDGCN__) &&                                                   \
    __has_builtin(__builtin_amdgcn_global_load_async_to_lds_b128) &&         \
    __has_builtin(__builtin_amdgcn_s_wait_asynccnt)
#define USE_ASYNC_LDS 1
#else
#define USE_ASYNC_LDS 0
#endif

static constexpr int BATCH = 4;
static constexpr int SEQ   = 4096;
static constexpr int EMB   = 1024;
static constexpr int HEADS = 16;
static constexpr int HDIM  = 64;

// 16B global -> LDS copy: async DMA path if available, else through VGPRs.
__device__ __forceinline__ void copy16_g2l(const _Float16* __restrict__ g,
                                           _Float16* __restrict__ l) {
#if USE_ASYNC_LDS
    __builtin_amdgcn_global_load_async_to_lds_b128(
        (__attribute__((address_space(1))) v4i_gcc*)g,
        (__attribute__((address_space(3))) v4i_gcc*)l, 0, 0);
#else
    *(uint4*)l = *(const uint4*)g;
#endif
}

template <int N>
__device__ __forceinline__ void wait_async() {
#if USE_ASYNC_LDS
    __builtin_amdgcn_s_wait_asynccnt(N);
#endif
}

// ---------------------------------------------------------------------------
// Fragment helpers. ISA 7.12.2 layouts (wave32):
//  A (16x32, f16): lane l -> row m=l&15, hi=l>>4.
//     v16h elems e=0..7  hold K = hi*8 + e          (contiguous 16B)
//               e=8..15 hold K = 16 + hi*8 + (e-8)  (contiguous 16B)
//  B (32x16, f16): lane l -> col n=l&15, hi=l>>4; elems hold K as for A.
//  C/D (16x16, f32): v8f elem r -> M = r + (l>>4)*8, N = l&15.
// ---------------------------------------------------------------------------

__device__ __forceinline__ v16h load_a_frag(const _Float16* __restrict__ sm,
                                            int row_base, int ld) {
    const int l  = threadIdx.x & 31;
    const int m  = l & 15;
    const int hi = l >> 4;
    const _Float16* r = sm + (row_base + m) * ld;
    v16h a;
#pragma unroll
    for (int e = 0; e < 8; ++e) a[e]     = r[hi * 8 + e];
#pragma unroll
    for (int e = 0; e < 8; ++e) a[8 + e] = r[16 + hi * 8 + e];
    return a;
}

__device__ __forceinline__ v16h load_b_frag(const _Float16* __restrict__ sm,
                                            int col_base, int ld) {
    const int l  = threadIdx.x & 31;
    const int n  = l & 15;
    const int hi = l >> 4;
    const _Float16* r = sm + col_base + n;
    v16h b;
#pragma unroll
    for (int e = 0; e < 8; ++e) b[e]     = r[(hi * 8 + e) * ld];
#pragma unroll
    for (int e = 0; e < 8; ++e) b[8 + e] = r[(16 + hi * 8 + e) * ld];
    return b;
}

// Activation modes: 0 = identity, 1 = elu(x)+1, 2 = sigmoid
template <int MODE>
__device__ __forceinline__ float activate(float v) {
    if constexpr (MODE == 1) return v > 0.0f ? v + 1.0f : __expf(v);
    if constexpr (MODE == 2) return 1.0f / (1.0f + __expf(-v));
    return v;
}

// ---------------------------------------------------------------------------
// Big GEMM: C[M,N] = act(A[M,K] * W[K,N] + bias[N]).  M%128==N%128==K%32==0.
// Macro-tile 128x128, BK=32, double-buffered async K pipeline.
// 8 waves: wave (wm=w>>1, wn=w&1) owns a 32x64 region = 8 accumulators.
// ---------------------------------------------------------------------------
template <int MODE, typename OutT>
__global__ __launch_bounds__(256, 2)
void gemm_wmma_kernel(const _Float16* __restrict__ A,
                      const _Float16* __restrict__ W,
                      const float* __restrict__ bias,
                      OutT* __restrict__ C,
                      int M, int N, int K) {
    __shared__ _Float16 As[2][128 * 32];
    __shared__ _Float16 Bs[2][32 * 128];

    const int tid  = threadIdx.x;
    const int wave = tid >> 5;
    const int lane = tid & 31;
    const int wm   = wave >> 1;         // 0..3
    const int wn   = wave & 1;          // 0..1
    const int m0   = blockIdx.y * 128;
    const int n0   = blockIdx.x * 128;

    v8f acc[2][4];
#pragma unroll
    for (int i = 0; i < 2; ++i)
#pragma unroll
        for (int j = 0; j < 4; ++j) acc[i][j] = (v8f){};

    // 4 async b128 ops per thread per tile (2 for A slab, 2 for B slab).
    auto issue_tile = [&](int buf, int k0) {
#pragma unroll
        for (int ii = 0; ii < 2; ++ii) {
            const int idx = tid + ii * 256;
            {   // A tile: 128 rows x 32 cols
                const int r = idx >> 2, c = (idx & 3) * 8;
                copy16_g2l(&A[(size_t)(m0 + r) * K + k0 + c],
                           &As[buf][r * 32 + c]);
            }
            {   // B tile: 32 rows x 128 cols
                const int r = idx >> 4, c = (idx & 15) * 8;
                copy16_g2l(&W[(size_t)(k0 + r) * N + n0 + c],
                           &Bs[buf][r * 128 + c]);
            }
        }
    };

    issue_tile(0, 0);
    int cur = 0;
    for (int k0 = 0; k0 < K; k0 += 32) {
        if (k0 + 32 < K) {
            issue_tile(cur ^ 1, k0 + 32);   // overlap: stream next slab
            wait_async<4>();                // oldest 4 (current slab) landed
        } else {
            wait_async<0>();
        }
        __syncthreads();

        const _Float16* as = As[cur];
        const _Float16* bs = Bs[cur];
        v16h af[2];
        af[0] = load_a_frag(as, wm * 32, 32);
        af[1] = load_a_frag(as, wm * 32 + 16, 32);
#pragma unroll
        for (int j = 0; j < 4; ++j) {
            const v16h bf = load_b_frag(bs, wn * 64 + j * 16, 128);
            acc[0][j] = WMMA_F16(af[0], bf, acc[0][j]);
            acc[1][j] = WMMA_F16(af[1], bf, acc[1][j]);
        }
        __syncthreads();   // everyone done with buf[cur] before it is refilled
        cur ^= 1;
    }

    // Epilogue: C layout -> M = r + (lane>>4)*8, N = lane&15
    const int nlo = lane & 15;
    const int mhi = (lane >> 4) * 8;
#pragma unroll
    for (int i = 0; i < 2; ++i) {
#pragma unroll
        for (int j = 0; j < 4; ++j) {
            const int col = n0 + wn * 64 + j * 16 + nlo;
            const float bv = bias ? bias[col] : 0.0f;
#pragma unroll
            for (int r = 0; r < 8; ++r) {
                const int row = m0 + wm * 32 + i * 16 + mhi + r;
                const float v = activate<MODE>(acc[i][j][r] + bv);
                C[(size_t)row * N + col] = (OutT)v;
            }
        }
    }
}

// ---------------------------------------------------------------------------
// KV kernel: one block per (b,h).  KV[64,64] = sum_s K[s,:]^T V[s,:]
// Double-buffered 32x64 K/V chunks in LDS; A-frag is a transposed LDS gather.
// 16 output tiles of 16x16 across 8 waves (2 tiles/wave), 128 K-steps.
// ---------------------------------------------------------------------------
__global__ __launch_bounds__(256, 2)
void kv_wmma_kernel(const _Float16* __restrict__ Kh,
                    const _Float16* __restrict__ Vh,
                    float* __restrict__ KV) {
    __shared__ _Float16 Ks[2][32 * 64];
    __shared__ _Float16 Vs[2][32 * 64];

    const int tid  = threadIdx.x;
    const int wave = tid >> 5;
    const int lane = tid & 31;
    const int bh   = blockIdx.x;
    const int b    = bh >> 4;
    const int h    = bh & 15;
    const size_t rowbase = (size_t)b * SEQ;
    const int colbase = h * HDIM;

    v8f acc[2];
    acc[0] = (v8f){};
    acc[1] = (v8f){};

    const int m  = lane & 15;
    const int hi = lane >> 4;
    const int lr = tid >> 3;              // 2048 f16 = 256 uint4/buffer
    const int lc = (tid & 7) * 8;

    auto issue_chunk = [&](int buf, int s0) {
        const size_t gofs = (rowbase + s0 + lr) * EMB + colbase + lc;
        copy16_g2l(&Kh[gofs], &Ks[buf][lr * 64 + lc]);
        copy16_g2l(&Vh[gofs], &Vs[buf][lr * 64 + lc]);
    };

    issue_chunk(0, 0);
    int cur = 0;
    for (int s0 = 0; s0 < SEQ; s0 += 32) {
        if (s0 + 32 < SEQ) {
            issue_chunk(cur ^ 1, s0 + 32);
            wait_async<2>();
        } else {
            wait_async<0>();
        }
        __syncthreads();

#pragma unroll
        for (int t = 0; t < 2; ++t) {
            const int tile = wave * 2 + t;
            const int ti = tile >> 2;   // d block (rows of KV)
            const int tj = tile & 3;    // e block (cols of KV)
            // A = K^T: A[m=d][k=s] = Ks[s][ti*16+m]  (transposed LDS gather)
            v16h af;
#pragma unroll
            for (int e = 0; e < 8; ++e)
                af[e] = Ks[cur][(hi * 8 + e) * 64 + ti * 16 + m];
#pragma unroll
            for (int e = 0; e < 8; ++e)
                af[8 + e] = Ks[cur][(16 + hi * 8 + e) * 64 + ti * 16 + m];
            // B = V chunk: lane = e col, elems = s rows
            const v16h bf = load_b_frag(Vs[cur], tj * 16, 64);
            acc[t] = WMMA_F16(af, bf, acc[t]);
        }
        __syncthreads();
        cur ^= 1;
    }

    const int nlo = lane & 15;
    const int mhi = (lane >> 4) * 8;
#pragma unroll
    for (int t = 0; t < 2; ++t) {
        const int tile = wave * 2 + t;
        const int ti = tile >> 2, tj = tile & 3;
#pragma unroll
        for (int r = 0; r < 8; ++r) {
            const int d = ti * 16 + mhi + r;
            const int e = tj * 16 + nlo;
            KV[(size_t)bh * HDIM * HDIM + d * HDIM + e] = acc[t][r];
        }
    }
}

// ---------------------------------------------------------------------------
// attn = Q @ KV per (b,h), fused with sigmoid-gate multiply.
// Block = (bh, s-tile of 128 rows). Wave w owns rows [16w,16w+16), 4 N-tiles.
// ---------------------------------------------------------------------------
__global__ __launch_bounds__(256, 2)
void attn_gate_kernel(const _Float16* __restrict__ Qh,
                      const float* __restrict__ KV,
                      const _Float16* __restrict__ Gh,
                      _Float16* __restrict__ AG) {
    __shared__ _Float16 Qs[128 * 64];
    __shared__ _Float16 KVs[64 * 64];

    const int tid  = threadIdx.x;
    const int wave = tid >> 5;
    const int lane = tid & 31;
    const int bh   = blockIdx.x;
    const int b    = bh >> 4;
    const int h    = bh & 15;
    const int s0   = blockIdx.y * 128;
    const size_t rowbase = (size_t)b * SEQ + s0;
    const int colbase = h * HDIM;

    // Q tile: 8192 f16 = 1024 uint4 -> 4 async copies per thread
#pragma unroll
    for (int ii = 0; ii < 4; ++ii) {
        const int idx = tid + ii * 256;
        const int r = idx >> 3, c = (idx & 7) * 8;
        copy16_g2l(&Qh[(rowbase + r) * EMB + colbase + c], &Qs[r * 64 + c]);
    }
    // KV tile: 4096 f32 -> f16 conversion (must pass through VALU)
#pragma unroll
    for (int ii = 0; ii < 16; ++ii) {
        const int idx = tid + ii * 256;
        KVs[idx] = (_Float16)KV[(size_t)bh * HDIM * HDIM + idx];
    }
    wait_async<0>();
    __syncthreads();

    v8f acc[4];
#pragma unroll
    for (int j = 0; j < 4; ++j) acc[j] = (v8f){};

#pragma unroll
    for (int kc = 0; kc < 2; ++kc) {
        const v16h af = load_a_frag(Qs + kc * 32, wave * 16, 64);
#pragma unroll
        for (int j = 0; j < 4; ++j) {
            const v16h bf = load_b_frag(KVs + kc * 32 * 64, j * 16, 64);
            acc[j] = WMMA_F16(af, bf, acc[j]);
        }
    }

    const int nlo = lane & 15;
    const int mhi = (lane >> 4) * 8;
#pragma unroll
    for (int j = 0; j < 4; ++j) {
#pragma unroll
        for (int r = 0; r < 8; ++r) {
            const int row = mhi + r + wave * 16;
            const size_t gi = (rowbase + row) * EMB + colbase + j * 16 + nlo;
            const float g = (float)Gh[gi];
            AG[gi] = (_Float16)(g * acc[j][r]);
        }
    }
}

// ---------------------------------------------------------------------------
// f32 -> f16 conversion (grid-stride)
// ---------------------------------------------------------------------------
__global__ void cvt_f32_to_f16(const float* __restrict__ in,
                               _Float16* __restrict__ out, size_t n) {
    for (size_t i = (size_t)blockIdx.x * blockDim.x + threadIdx.x; i < n;
         i += (size_t)gridDim.x * blockDim.x)
        out[i] = (_Float16)in[i];
}

// ---------------------------------------------------------------------------
extern "C" void kernel_launch(void* const* d_in, const int* in_sizes, int n_in,
                              void* d_out, int out_size, void* d_ws, size_t ws_size,
                              hipStream_t stream) {
    const float* x  = (const float*)d_in[0];
    const float* Wq = (const float*)d_in[1];
    const float* bq = (const float*)d_in[2];
    const float* Wk = (const float*)d_in[3];
    const float* bk = (const float*)d_in[4];
    const float* Wv = (const float*)d_in[5];
    const float* bv = (const float*)d_in[6];
    const float* Wg = (const float*)d_in[7];
    const float* bg = (const float*)d_in[8];
    const float* Wo = (const float*)d_in[9];
    const float* bo = (const float*)d_in[10];
    float* out = (float*)d_out;

    const size_t Mr = (size_t)BATCH * SEQ;       // 16384
    char* ws = (char*)d_ws;
    size_t off = 0;
    auto take = [&](size_t bytes) {
        char* p = ws + off;
        off += (bytes + 255) & ~(size_t)255;
        return p;
    };

    _Float16* xh  = (_Float16*)take(Mr * EMB * 2);
    _Float16* wqh = (_Float16*)take((size_t)EMB * EMB * 2);
    _Float16* wkh = (_Float16*)take((size_t)EMB * EMB * 2);
    _Float16* wvh = (_Float16*)take((size_t)EMB * EMB * 2);
    _Float16* wgh = (_Float16*)take((size_t)EMB * EMB * 2);
    _Float16* woh = (_Float16*)take((size_t)EMB * EMB * 2);
    _Float16* qh  = (_Float16*)take(Mr * EMB * 2);
    _Float16* kh  = (_Float16*)take(Mr * EMB * 2);
    _Float16* vh  = (_Float16*)take(Mr * EMB * 2);
    _Float16* gh  = (_Float16*)take(Mr * EMB * 2);
    _Float16* agh = (_Float16*)take(Mr * EMB * 2);
    float*    kvb = (float*)take((size_t)BATCH * HEADS * HDIM * HDIM * 4);
    (void)ws_size; (void)n_in; (void)in_sizes; (void)out_size;

    // 1) f32 -> f16 conversions
    cvt_f32_to_f16<<<2048, 256, 0, stream>>>(x,  xh,  Mr * EMB);
    cvt_f32_to_f16<<<1024, 256, 0, stream>>>(Wq, wqh, (size_t)EMB * EMB);
    cvt_f32_to_f16<<<1024, 256, 0, stream>>>(Wk, wkh, (size_t)EMB * EMB);
    cvt_f32_to_f16<<<1024, 256, 0, stream>>>(Wv, wvh, (size_t)EMB * EMB);
    cvt_f32_to_f16<<<1024, 256, 0, stream>>>(Wg, wgh, (size_t)EMB * EMB);
    cvt_f32_to_f16<<<1024, 256, 0, stream>>>(Wo, woh, (size_t)EMB * EMB);

    // 2) Projections: grid (N/128, M/128) = (8, 128)
    const dim3 ggrid(EMB / 128, (int)(Mr / 128));
    gemm_wmma_kernel<1, _Float16><<<ggrid, 256, 0, stream>>>(xh, wqh, bq, qh,
                                                             (int)Mr, EMB, EMB);
    gemm_wmma_kernel<1, _Float16><<<ggrid, 256, 0, stream>>>(xh, wkh, bk, kh,
                                                             (int)Mr, EMB, EMB);
    gemm_wmma_kernel<0, _Float16><<<ggrid, 256, 0, stream>>>(xh, wvh, bv, vh,
                                                             (int)Mr, EMB, EMB);
    gemm_wmma_kernel<2, _Float16><<<ggrid, 256, 0, stream>>>(xh, wgh, bg, gh,
                                                             (int)Mr, EMB, EMB);

    // 3) KV = phi(K)^T V per (b,h)
    kv_wmma_kernel<<<BATCH * HEADS, 256, 0, stream>>>(kh, vh, kvb);

    // 4) attn = phi(Q) @ KV, fused gate multiply
    attn_gate_kernel<<<dim3(BATCH * HEADS, SEQ / 128), 256, 0, stream>>>(qh, kvb,
                                                                         gh, agh);

    // 5) out = (gate*attn) @ Wo + bo  (f32 output)
    gemm_wmma_kernel<0, float><<<ggrid, 256, 0, stream>>>(agh, woh, bo, out,
                                                          (int)Mr, EMB, EMB);
}
